// DeconvBlock_76347338653931
// MI455X (gfx1250) — matrix operand order repaired and checked by reading, not compiled, not run on hardware
//
#include <hip/hip_runtime.h>
#include <hip/hip_bf16.h>
#include <math.h>

// ---------------------------------------------------------------------------
// CDNA5 / gfx1250 implementation of the DeconvBlock pipeline.
// wave32, WMMA 16x16x32 f16 with f32 accumulate.
// Deconv stage uses GLOBAL_LOAD_ASYNC_TO_LDS_B32 (ASYNCcnt) for A staging.
// ---------------------------------------------------------------------------

typedef _Float16 f16_t;
typedef __attribute__((ext_vector_type(16))) _Float16 v16h;
typedef __attribute__((ext_vector_type(8)))  _Float16 v8h;
typedef __attribute__((ext_vector_type(8)))  float    v8f;

#define B_   16
#define C_   256
#define CO_  256
#define H_   32
#define W_   32
#define HW_  1024           // 32*32
#define NPIX 16384          // B*H*W
#define KDEF 2304           // C*9
#define OH_  64
#define OW_  64

// ---------------------------------------------------------------------------
// Weight prep: w_def [Co,C,3,3] fp32 -> f16 [co][k] with k = tap*256 + c
// ---------------------------------------------------------------------------
__global__ void k_prep_wdefT(const float* __restrict__ w_def, f16_t* __restrict__ wdefT) {
    int idx = blockIdx.x * 256 + threadIdx.x;           // 589824 total
    if (idx >= CO_ * KDEF) return;
    int co = idx / KDEF;
    int k  = idx - co * KDEF;
    int tap = k >> 8;
    int c   = k & 255;
    wdefT[idx] = (f16_t)w_def[(size_t)co * KDEF + c * 9 + tap];
}

// ---------------------------------------------------------------------------
// Weight prep for transpose conv, 4 parity classes (dy,dx):
//   wdcT[((cls*256+co)<<10) + tt*256 + ci] = w_dc[ci][co][kh(dy,tt)][kw(dx,tt)]
// ---------------------------------------------------------------------------
__global__ void k_prep_wdcT(const float* __restrict__ w_dc, f16_t* __restrict__ wdcT) {
    int idx = blockIdx.x * 256 + threadIdx.x;           // 1048576 total
    int cls = idx >> 18;
    int rem = idx & 262143;
    int co  = rem >> 10;
    int k   = rem & 1023;
    int tt  = k >> 8;
    int ci  = k & 255;
    int dy = cls >> 1, dx = cls & 1;
    int kh = (dy == 0) ? ((tt & 2) ? 3 : 1) : ((tt & 2) ? 2 : 0);
    int kw = (dx == 0) ? ((tt & 1) ? 3 : 1) : ((tt & 1) ? 2 : 0);
    wdcT[idx] = (f16_t)w_dc[(((size_t)ci * 256 + co) * 4 + kh) * 4 + kw];
}

// ---------------------------------------------------------------------------
// Offset conv: plain 3x3 conv, pad 1, 18 output channels (small; scalar MACs)
// out layout [b][oc][h][w]
// ---------------------------------------------------------------------------
__global__ void k_offset_conv(const float* __restrict__ x,
                              const float* __restrict__ w_off,
                              const float* __restrict__ b_off,
                              float* __restrict__ off_buf) {
    int idx = blockIdx.x * 256 + threadIdx.x;           // 294912 total
    int b   = idx / (18 * HW_);
    int rem = idx - b * 18 * HW_;
    int oc  = rem >> 10;
    int sp  = rem & 1023;
    int h = sp >> 5, w = sp & 31;

    int  tindex[9];
    unsigned tok = 0;
    #pragma unroll
    for (int tap = 0; tap < 9; ++tap) {
        int yy = h + tap / 3 - 1;
        int xx = w + tap % 3 - 1;
        bool ok = (unsigned)yy < 32u && (unsigned)xx < 32u;
        tindex[tap] = (yy << 5) + xx;
        if (ok) tok |= 1u << tap;
    }
    float acc = b_off[oc];
    const float* wrow = w_off + (size_t)oc * KDEF;
    for (int c = 0; c < C_; ++c) {
        const float* xp = x + ((size_t)(b * C_ + c) << 10);
        const float* wp = wrow + c * 9;
        #pragma unroll
        for (int tap = 0; tap < 9; ++tap)
            if (tok & (1u << tap)) acc += xp[tindex[tap]] * wp[tap];
    }
    off_buf[((size_t)(b * 18 + oc) << 10) + sp] = acc;
}

// ---------------------------------------------------------------------------
// Deformable conv as WMMA GEMM.
// Per WG: 16 pixels (M), 8 waves x 32 co (N=256), K loop over 72 chunks of 32.
// ---------------------------------------------------------------------------
__global__ void __launch_bounds__(256)
k_deform_gemm(const float* __restrict__ x,
              const float* __restrict__ off_buf,
              const f16_t* __restrict__ wdefT,
              const float* __restrict__ b_def,
              float* __restrict__ y1) {
    __shared__ int   sidx[16 * 9 * 4];
    __shared__ float swt [16 * 9 * 4];
    __shared__ f16_t Atile[16 * 32];

    const int tid  = threadIdx.x;
    const int lane = tid & 31;
    const int wid  = tid >> 5;

    // ---- phase 1: bilinear corner indices/weights for 16 px x 9 taps ------
    if (tid < 144) {
        int m = tid / 9, tap = tid % 9;
        int pixel = blockIdx.x * 16 + m;
        int b = pixel >> 10, rem = pixel & 1023;
        int h = rem >> 5, w = rem & 31;
        const float* offp = off_buf + ((size_t)(b * 18 + 2 * tap) << 10) + (h << 5) + w;
        float dy = offp[0];
        float dx = offp[1 << 10];
        float py = (float)(h + tap / 3 - 1) + dy;
        float px = (float)(w + tap % 3 - 1) + dx;
        float y0f = floorf(py), x0f = floorf(px);
        float wy = py - y0f,    wx = px - x0f;
        int y0 = (int)y0f, x0 = (int)x0f;
        int base = (m * 9 + tap) * 4;
        #pragma unroll
        for (int cr = 0; cr < 4; ++cr) {
            int ty = cr >> 1, tx = cr & 1;
            int yi = y0 + ty, xi = x0 + tx;
            bool valid = (unsigned)yi < 32u && (unsigned)xi < 32u;
            float wgt = (ty ? wy : 1.f - wy) * (tx ? wx : 1.f - wx);
            swt[base + cr] = valid ? wgt : 0.f;
            int yc = yi < 0 ? 0 : (yi > 31 ? 31 : yi);
            int xc = xi < 0 ? 0 : (xi > 31 ? 31 : xi);
            sidx[base + cr] = (yc << 5) + xc;
        }
    }
    __syncthreads();

    const int mst   = tid >> 4;            // staging row
    const int ccst  = (tid & 15) * 2;      // staging channel pair
    const int pixst = blockIdx.x * 16 + mst;
    const int bst   = pixst >> 10;

    const int mrow  = lane & 15;
    const int abase = (lane & 16) ? 16 : 0;
    const int bbase = (lane & 16) ? 16 : 0;

    v8f acc0 = {};
    v8f acc1 = {};

    for (int chunk = 0; chunk < 72; ++chunk) {
        const int tap = chunk >> 3;
        const int c0  = (chunk & 7) * 32;
        __syncthreads();                   // previous fragments consumed
        // ---- stage A: gather bilinear samples for this (tap, c0..c0+31) ---
        #pragma unroll
        for (int u = 0; u < 2; ++u) {
            int c = c0 + ccst + u;
            const float* xp = x + ((size_t)(bst * C_ + c) << 10);
            int mi = (mst * 9 + tap) * 4;
            float v = swt[mi + 0] * xp[sidx[mi + 0]]
                    + swt[mi + 1] * xp[sidx[mi + 1]]
                    + swt[mi + 2] * xp[sidx[mi + 2]]
                    + swt[mi + 3] * xp[sidx[mi + 3]];
            Atile[mst * 32 + ccst + u] = (f16_t)v;
        }
        __syncthreads();

        // ---- fragments + WMMA ---------------------------------------------
        v8h lo = *(const v8h*)&Atile[mrow * 32 + abase];
        v8h hi = *(const v8h*)&Atile[mrow * 32 + abase + 8];
        v16h av = __builtin_shufflevector(lo, hi, 0,1,2,3,4,5,6,7,8,9,10,11,12,13,14,15);

        int co0 = wid * 32 + (lane & 15);
        int co1 = co0 + 16;
        const f16_t* bp0 = &wdefT[(size_t)co0 * KDEF + tap * 256 + c0 + bbase];
        const f16_t* bp1 = &wdefT[(size_t)co1 * KDEF + tap * 256 + c0 + bbase];
        v16h bv0 = *(const v16h*)bp0;
        v16h bv1 = *(const v16h*)bp1;
        if (chunk + 1 < 72) {
            int ntap = (chunk + 1) >> 3, nc0 = ((chunk + 1) & 7) * 32;
            __builtin_prefetch(&wdefT[(size_t)co0 * KDEF + ntap * 256 + nc0], 0, 3);
        }
        acc0 = __builtin_amdgcn_wmma_f32_16x16x32_f16(false, av, false, bv0, (short)0, acc0, false, false);
        acc1 = __builtin_amdgcn_wmma_f32_16x16x32_f16(false, av, false, bv1, (short)0, acc1, false, false);
    }

    // ---- epilogue: y1[pixel][co] = acc + b_def ----------------------------
    const int Moff = (lane & 16) ? 8 : 0;
    #pragma unroll
    for (int r = 0; r < 8; ++r) {
        int M = r + Moff;
        int pixel = blockIdx.x * 16 + M;
        int coA = wid * 32 + (lane & 15);
        int coB = coA + 16;
        y1[(size_t)pixel * CO_ + coA] = acc0[r] + b_def[coA];
        y1[(size_t)pixel * CO_ + coB] = acc1[r] + b_def[coB];
    }
}

// ---------------------------------------------------------------------------
// BN1 stats: one WG per channel over 16384 pixels; fold gamma/beta.
// ---------------------------------------------------------------------------
__global__ void k_bn_stats1(const float* __restrict__ y1,
                            const float* __restrict__ gamma,
                            const float* __restrict__ beta,
                            float* __restrict__ ss) {
    int co = blockIdx.x, tid = threadIdx.x;
    float s = 0.f, q = 0.f;
    for (int p = tid; p < NPIX; p += 256) {
        float v = y1[(size_t)p * CO_ + co];
        s += v; q += v * v;
    }
    __shared__ float rs[256], rq[256];
    rs[tid] = s; rq[tid] = q;
    __syncthreads();
    for (int off = 128; off > 0; off >>= 1) {
        if (tid < off) { rs[tid] += rs[tid + off]; rq[tid] += rq[tid + off]; }
        __syncthreads();
    }
    if (tid == 0) {
        float mean = rs[0] * (1.f / NPIX);
        float var  = rq[0] * (1.f / NPIX) - mean * mean;
        float rstd = rsqrtf(var + 1e-5f);
        float sc = gamma[co] * rstd;
        ss[co * 2]     = sc;
        ss[co * 2 + 1] = beta[co] - mean * sc;
    }
}

// BN1 apply + SiLU, convert to f16 activation [pixel][co]
__global__ void k_bn_apply1(const float* __restrict__ y1,
                            const float* __restrict__ ss,
                            f16_t* __restrict__ y1act) {
    int i = blockIdx.x * 256 + threadIdx.x;     // 4194304 total
    int co = i & 255;
    float y = ss[co * 2] * y1[i] + ss[co * 2 + 1];
    float sg = 1.f / (1.f + __expf(-y));
    y1act[i] = (f16_t)(y * sg);
}

// ---------------------------------------------------------------------------
// ConvTranspose2d(4x4, s=2, p=1) as 4 parity-class WMMA GEMMs.
// WG: cls, batch b, input row a, 16-col segment t.  K = 4 taps * 256 ch.
// A staging uses GLOBAL_LOAD_ASYNC_TO_LDS_B32 (ASYNCcnt) for in-range lanes,
// ds_store zero-fill for edge lanes.  Writes pre-BN fp32 into d_out.
// ---------------------------------------------------------------------------
__global__ void __launch_bounds__(256)
k_deconv_gemm(const f16_t* __restrict__ y1act,
              const f16_t* __restrict__ wdcT,
              float* __restrict__ out) {
    __shared__ f16_t Atile[16 * 32];

    const int wg  = blockIdx.x;
    const int cls = wg >> 10;
    const int rem = wg & 1023;
    const int b   = rem >> 6;
    const int r2  = rem & 63;
    const int a   = r2 >> 1;
    const int t   = r2 & 1;
    const int dy = cls >> 1, dx = cls & 1;

    const int tid  = threadIdx.x;
    const int lane = tid & 31;
    const int wid  = tid >> 5;

    const int mst  = tid >> 4;
    const int ccst = (tid & 15) * 2;
    // low 32 bits of the generic LDS pointer == wave-relative LDS byte address
    const unsigned ldsOff = (unsigned)(uintptr_t)(&Atile[mst * 32 + ccst]);

    const int mrow  = lane & 15;
    const int abase = (lane & 16) ? 16 : 0;
    const int bbase = (lane & 16) ? 16 : 0;

    v8f acc0 = {};
    v8f acc1 = {};

    for (int chunk = 0; chunk < 32; ++chunk) {
        const int tt = chunk >> 3;
        const int c0 = (chunk & 7) * 32;
        const int dih = (dy == 0) ? ((tt & 2) ? -1 : 0) : ((tt & 2) ? 0 : 1);
        const int diw = (dx == 0) ? ((tt & 1) ? -1 : 0) : ((tt & 1) ? 0 : 1);

        __syncthreads();
        // ---- stage A: async copy (valid lanes) / zero-fill (edge lanes) ---
        {
            int ih = a + dih;
            int iw = t * 16 + mst + diw;
            bool ok = (unsigned)ih < 32u && (unsigned)iw < 32u;
            if (ok) {
                int pixelIn = (b << 10) + (ih << 5) + iw;
                unsigned long long ga =
                    (unsigned long long)(uintptr_t)&y1act[((size_t)pixelIn << 8) + c0 + ccst];
                asm volatile("global_load_async_to_lds_b32 %0, %1, off"
                             :: "v"(ldsOff), "v"(ga)
                             : "memory");
            } else {
                *(unsigned*)&Atile[mst * 32 + ccst] = 0u;
            }
        }
        asm volatile("s_wait_asynccnt 0x0" ::: "memory");
        __syncthreads();

        v8h lo = *(const v8h*)&Atile[mrow * 32 + abase];
        v8h hi = *(const v8h*)&Atile[mrow * 32 + abase + 8];
        v16h av = __builtin_shufflevector(lo, hi, 0,1,2,3,4,5,6,7,8,9,10,11,12,13,14,15);

        int co0 = wid * 32 + (lane & 15);
        int co1 = co0 + 16;
        v16h bv0 = *(const v16h*)&wdcT[((size_t)((cls << 8) + co0) << 10) + tt * 256 + c0 + bbase];
        v16h bv1 = *(const v16h*)&wdcT[((size_t)((cls << 8) + co1) << 10) + tt * 256 + c0 + bbase];
        if (chunk + 1 < 32) {
            int ntt = (chunk + 1) >> 3, nc0 = ((chunk + 1) & 7) * 32;
            __builtin_prefetch(&wdcT[((size_t)((cls << 8) + co0) << 10) + ntt * 256 + nc0], 0, 3);
        }
        acc0 = __builtin_amdgcn_wmma_f32_16x16x32_f16(false, av, false, bv0, (short)0, acc0, false, false);
        acc1 = __builtin_amdgcn_wmma_f32_16x16x32_f16(false, av, false, bv1, (short)0, acc1, false, false);
    }

    // ---- epilogue: scatter into [b][co][oh][ow] ---------------------------
    const int Moff = (lane & 16) ? 8 : 0;
    const int oh = 2 * a + dy;
    #pragma unroll
    for (int r = 0; r < 8; ++r) {
        int M = r + Moff;
        int ow = 2 * (t * 16 + M) + dx;
        int coA = wid * 32 + (lane & 15);
        int coB = coA + 16;
        out[((size_t)((b << 8) + coA) << 12) + (oh << 6) + ow] = acc0[r];
        out[((size_t)((b << 8) + coB) << 12) + (oh << 6) + ow] = acc1[r];
    }
}

// ---------------------------------------------------------------------------
// BN2 stats over [b][co][64][64] (65536 samples/channel), fold gamma/beta.
// ---------------------------------------------------------------------------
__global__ void k_bn_stats2(const float* __restrict__ out,
                            const float* __restrict__ gamma,
                            const float* __restrict__ beta,
                            float* __restrict__ ss) {
    int co = blockIdx.x, tid = threadIdx.x;
    float s = 0.f, q = 0.f;
    for (int i = tid; i < 65536; i += 256) {
        int b = i >> 12, sp = i & 4095;
        float v = out[((size_t)((b << 8) + co) << 12) + sp];
        s += v; q += v * v;
    }
    __shared__ float rs[256], rq[256];
    rs[tid] = s; rq[tid] = q;
    __syncthreads();
    for (int off = 128; off > 0; off >>= 1) {
        if (tid < off) { rs[tid] += rs[tid + off]; rq[tid] += rq[tid + off]; }
        __syncthreads();
    }
    if (tid == 0) {
        float mean = rs[0] * (1.f / 65536.f);
        float var  = rq[0] * (1.f / 65536.f) - mean * mean;
        float rstd = rsqrtf(var + 1e-5f);
        float sc = gamma[co] * rstd;
        ss[co * 2]     = sc;
        ss[co * 2 + 1] = beta[co] - mean * sc;
    }
}

// BN2 apply + SiLU in place on d_out
__global__ void k_bn_apply2(float* __restrict__ out, const float* __restrict__ ss) {
    int i = blockIdx.x * 256 + threadIdx.x;     // 16777216 total
    int co = (i >> 12) & 255;
    float y = ss[co * 2] * out[i] + ss[co * 2 + 1];
    float sg = 1.f / (1.f + __expf(-y));
    out[i] = y * sg;
}

// ---------------------------------------------------------------------------
// Host launcher
// ---------------------------------------------------------------------------
static inline size_t align256(size_t v) { return (v + 255) & ~(size_t)255; }

extern "C" void kernel_launch(void* const* d_in, const int* in_sizes, int n_in,
                              void* d_out, int out_size, void* d_ws, size_t ws_size,
                              hipStream_t stream) {
    (void)in_sizes; (void)n_in; (void)out_size; (void)ws_size;

    const float* x      = (const float*)d_in[0];
    const float* w_off  = (const float*)d_in[1];
    const float* b_off  = (const float*)d_in[2];
    const float* w_def  = (const float*)d_in[3];
    const float* b_def  = (const float*)d_in[4];
    const float* gamma1 = (const float*)d_in[5];
    const float* beta1  = (const float*)d_in[6];
    const float* w_dc   = (const float*)d_in[7];
    const float* gamma2 = (const float*)d_in[8];
    const float* beta2  = (const float*)d_in[9];
    float* out = (float*)d_out;

    char* ws = (char*)d_ws;
    size_t off = 0;
    float* off_buf = (float*)(ws + off); off = align256(off + (size_t)B_ * 18 * HW_ * 4);
    f16_t* wdefT   = (f16_t*)(ws + off); off = align256(off + (size_t)CO_ * KDEF * 2);
    float* y1      = (float*)(ws + off); off = align256(off + (size_t)NPIX * CO_ * 4);
    f16_t* y1act   = (f16_t*)(ws + off); off = align256(off + (size_t)NPIX * CO_ * 2);
    f16_t* wdcT    = (f16_t*)(ws + off); off = align256(off + (size_t)4 * CO_ * 1024 * 2);
    float* ss1     = (float*)(ws + off); off = align256(off + 512 * 4);
    float* ss2     = (float*)(ws + off); off = align256(off + 512 * 4);

    // weight prep
    k_prep_wdefT<<<dim3((CO_ * KDEF + 255) / 256), dim3(256), 0, stream>>>(w_def, wdefT);
    k_prep_wdcT <<<dim3(4096),  dim3(256), 0, stream>>>(w_dc, wdcT);

    // stage 1: offset conv
    k_offset_conv<<<dim3(1152), dim3(256), 0, stream>>>(x, w_off, b_off, off_buf);

    // stage 2: deformable conv (WMMA GEMM)
    k_deform_gemm<<<dim3(NPIX / 16), dim3(256), 0, stream>>>(x, off_buf, wdefT, b_def, y1);

    // stage 3: BN1 + SiLU -> f16
    k_bn_stats1<<<dim3(CO_),   dim3(256), 0, stream>>>(y1, gamma1, beta1, ss1);
    k_bn_apply1<<<dim3(NPIX * CO_ / 256), dim3(256), 0, stream>>>(y1, ss1, y1act);

    // stage 4: conv transpose (4 parity-class WMMA GEMMs, async-to-LDS staging)
    k_deconv_gemm<<<dim3(4096), dim3(256), 0, stream>>>(y1act, wdcT, out);

    // stage 5: BN2 + SiLU in place
    k_bn_stats2<<<dim3(CO_), dim3(256), 0, stream>>>(out, gamma2, beta2, ss2);
    k_bn_apply2<<<dim3((size_t)B_ * CO_ * OH_ * OW_ / 256), dim3(256), 0, stream>>>(out, ss2);
}
